// BipartiteGNNConvFactorToVariable_15564961481300
// MI455X (gfx1250) — matrix operand order repaired and checked by reading, not compiled, not run on hardware
//
#include <hip/hip_runtime.h>

// ---------------------------------------------------------------------------
// BipartiteGNNConv: edge-MLP + segment_sum + combine-MLP, bf16 WMMA datapath
//   D = 128, K = 2D = 256; WMMA tile 16x16x32 bf16 -> f32 accum
//   Weights staged TRANSPOSED in LDS so every WMMA fragment is contiguous
//   (two ds_load_b128 per fragment, no scalar u16 gathers).
// ---------------------------------------------------------------------------

typedef __attribute__((ext_vector_type(16))) __bf16 v16bf;
typedef __attribute__((ext_vector_type(8)))  __bf16 v8bf;
typedef __attribute__((ext_vector_type(4)))  __bf16 v4bf;
typedef __attribute__((ext_vector_type(8)))  float  v8f;

#define NDIM 128            // feature dim / MLP output dim
#define KDIM 256            // concat dim (2*D)
#define KSTRIDE 264         // padded k-stride (bf16 elems) for LDS tiles
#define SMEM_ELEMS (NDIM * KSTRIDE + 16 * KSTRIDE)   // Wt[128][264] + A[16][264]
#define SMEM_BYTES (SMEM_ELEMS * 2)                  // 76032 B
#define TILES_PER_BLOCK 8   // 16-row tiles handled per block

// Concatenate two contiguous 8-elem bf16 chunks into one fragment register set.
__device__ __forceinline__ v16bf frag_2x8(const __bf16* p0, const __bf16* p1) {
    const v8bf lo = *(const v8bf*)p0;   // ds_load_b128
    const v8bf hi = *(const v8bf*)p1;   // ds_load_b128
    return __builtin_shufflevector(lo, hi, 0, 1, 2, 3, 4, 5, 6, 7,
                                           8, 9, 10, 11, 12, 13, 14, 15);
}

// A (16-bit, 16x32): lane (m = lane&15, h = lane>>4) holds, per VGPR pair,
// K = {h*8..h*8+7} and {16+h*8..16+h*8+7} -> two contiguous 8-elem runs.
__device__ __forceinline__ v16bf load_a_frag(const __bf16* sA, int mrow,
                                             int half, int ks) {
    const __bf16* p = sA + mrow * KSTRIDE + ks * 32 + half * 8;
    return frag_2x8(p, p + 16);
}

// B (16-bit, 32x16): lane (n = lane&15, h = lane>>4) holds
// K = ks*32 + {h*8..h*8+7} and {16+h*8..16+h*8+7} -> two contiguous 8-elem runs
// in the TRANSPOSED weight tile Wt[n][k].
__device__ __forceinline__ v16bf load_b_frag(const __bf16* sWt, int ncol,
                                             int half, int ks) {
    const __bf16* p = sWt + ncol * KSTRIDE + ks * 32 + half * 8;
    return frag_2x8(p, p + 16);
}

// ---- pass 0: zero the segment-sum accumulator -----------------------------

__global__ void zero_f32_kernel(float* __restrict__ p, long long n) {
    long long i = (long long)blockIdx.x * blockDim.x + threadIdx.x;
    if (i < n) p[i] = 0.0f;
}

// ---- shared staging helpers ----------------------------------------------

// Stage W (f32 [K,N] row-major) transposed into LDS as bf16 Wt[n][k].
__device__ __forceinline__ void stage_weights_T(const float* __restrict__ W,
                                                __bf16* sWt, int tid) {
    for (int i = tid; i < KDIM * NDIM; i += 256) {
        const int k = i >> 7;            // row of W
        const int n = i & (NDIM - 1);    // col of W
        sWt[n * KSTRIDE + k] = (__bf16)W[i];   // coalesced read, strided LDS write
    }
}

// Convert 16 f32 -> bf16 and store as 4x 8-byte LDS stores.
__device__ __forceinline__ void stage_row16(const float* __restrict__ src,
                                            __bf16* dst) {
#pragma unroll
    for (int j = 0; j < 16; j += 4) {
        const float4 v = *(const float4*)(src + j);
        v4bf w;
        w[0] = (__bf16)v.x; w[1] = (__bf16)v.y;
        w[2] = (__bf16)v.z; w[3] = (__bf16)v.w;
        *(v4bf*)(dst + j) = w;           // ds_store_b64
    }
}

// ---- pass 1: edge MLP + atomic segment_sum --------------------------------
// msg = relu([variables[s], factors[r]] @ Wm + bm); agg[s] += msg

__global__ __launch_bounds__(256)
void edge_msg_kernel(const float* __restrict__ variables,
                     const float* __restrict__ factors,
                     const int*  __restrict__ senders,
                     const int*  __restrict__ receivers,
                     const float* __restrict__ Wm,
                     const float* __restrict__ bm,
                     float* __restrict__ agg,
                     int E) {
    extern __shared__ __bf16 smem[];
    __bf16* sWt = smem;                   // [NDIM][KSTRIDE]  transposed weights
    __bf16* sA  = smem + NDIM * KSTRIDE;  // [16][KSTRIDE]    activation tile

    const int tid  = threadIdx.x;
    const int lane = tid & 31;
    const int wave = tid >> 5;            // 0..7 -> N tile [16w, 16w+16)
    const int half = lane >> 4;
    const int nl   = lane & 15;
    const int nG   = wave * 16 + nl;      // global output column

    stage_weights_T(Wm, sWt, tid);

    const float bias = bm[nG];
    const int row = tid >> 4;             // 0..15: edge row within tile
    const int seg = tid & 15;             // 16-element K segment

    for (int t = 0; t < TILES_PER_BLOCK; ++t) {
        const int eBase = (blockIdx.x * TILES_PER_BLOCK + t) * 16;
        if (eBase >= E) break;

        // prefetch next tile's index lists into cache (global_prefetch_b8)
        if (eBase + 16 < E) {
            __builtin_prefetch(senders + eBase + 16, 0, 0);
            __builtin_prefetch(receivers + eBase + 16, 0, 0);
        }
        __syncthreads();  // sWt ready (t==0) / previous tile's WMMA reads done

        // Gather A tile: 16 edges x 256 K, f32 -> bf16.
        {
            const int e = min(eBase + row, E - 1);
            const float* src;
            if (seg < 8) {
                const int s = senders[e];
                src = variables + (size_t)s * NDIM + seg * 16;
            } else {
                const int r = receivers[e];
                src = factors + (size_t)r * NDIM + (seg - 8) * 16;
            }
            stage_row16(src, sA + row * KSTRIDE + seg * 16);
        }
        __syncthreads();

        // 16x16 output tile, K = 256 in 8 steps of 32.
        v8f c = {};
#pragma unroll
        for (int ks = 0; ks < 8; ++ks) {
            const v16bf a = load_a_frag(sA, nl, half, ks);
            const v16bf b = load_b_frag(sWt, nG, half, ks);
            c = __builtin_amdgcn_wmma_f32_16x16x32_bf16(
                    false, a, false, b, (short)0, c, false, false);
        }

        // bias + ReLU + scatter-add (segment_sum over senders)
#pragma unroll
        for (int v = 0; v < 8; ++v) {
            const int m = v + half * 8;   // C layout: M = vgpr + 8*laneHalf
            const int e = eBase + m;
            float val = c[v] + bias;
            val = val > 0.0f ? val : 0.0f;
            if (e < E) {
                const int s = senders[e];
                atomicAdd(agg + (size_t)s * NDIM + nG, val);
            }
        }
    }
}

// ---- pass 2: combine MLP + residual ---------------------------------------
// out = variables + relu([variables, agg] @ Wc + bc)

__global__ __launch_bounds__(256)
void node_combine_kernel(const float* __restrict__ variables,
                         const float* __restrict__ agg,
                         const float* __restrict__ Wc,
                         const float* __restrict__ bc,
                         float* __restrict__ out,
                         int NVAR) {
    extern __shared__ __bf16 smem[];
    __bf16* sWt = smem;
    __bf16* sA  = smem + NDIM * KSTRIDE;

    const int tid  = threadIdx.x;
    const int lane = tid & 31;
    const int wave = tid >> 5;
    const int half = lane >> 4;
    const int nl   = lane & 15;
    const int nG   = wave * 16 + nl;

    stage_weights_T(Wc, sWt, tid);

    const float bias = bc[nG];
    const int row = tid >> 4;
    const int seg = tid & 15;

    for (int t = 0; t < TILES_PER_BLOCK; ++t) {
        const int vBase = (blockIdx.x * TILES_PER_BLOCK + t) * 16;
        if (vBase >= NVAR) break;
        __syncthreads();

        {
            const int node = min(vBase + row, NVAR - 1);
            const float* src = (seg < 8)
                ? variables + (size_t)node * NDIM + seg * 16
                : agg       + (size_t)node * NDIM + (seg - 8) * 16;
            stage_row16(src, sA + row * KSTRIDE + seg * 16);
        }
        __syncthreads();

        v8f c = {};
#pragma unroll
        for (int ks = 0; ks < 8; ++ks) {
            const v16bf a = load_a_frag(sA, nl, half, ks);
            const v16bf b = load_b_frag(sWt, nG, half, ks);
            c = __builtin_amdgcn_wmma_f32_16x16x32_bf16(
                    false, a, false, b, (short)0, c, false, false);
        }

#pragma unroll
        for (int v = 0; v < 8; ++v) {
            const int m = v + half * 8;
            const int node = vBase + m;
            float val = c[v] + bias;
            val = val > 0.0f ? val : 0.0f;
            if (node < NVAR) {
                const size_t idx = (size_t)node * NDIM + nG;
                out[idx] = variables[idx] + val;
            }
        }
    }
}

// ---------------------------------------------------------------------------

extern "C" void kernel_launch(void* const* d_in, const int* in_sizes, int n_in,
                              void* d_out, int out_size, void* d_ws, size_t ws_size,
                              hipStream_t stream) {
    const float* variables = (const float*)d_in[0];
    const float* factors   = (const float*)d_in[1];
    const int*   senders   = (const int*)  d_in[2];
    const int*   receivers = (const int*)  d_in[3];
    const float* Wm        = (const float*)d_in[4];
    const float* bm        = (const float*)d_in[5];
    const float* Wc        = (const float*)d_in[6];
    const float* bc        = (const float*)d_in[7];
    float*       out       = (float*)d_out;
    float*       agg       = (float*)d_ws;   // [NVAR, 128] f32 accumulator

    const int NVAR = in_sizes[0] / NDIM;
    const int E    = in_sizes[2];

    // Allow >64KB dynamic LDS (320KB/WGP on CDNA5). Not a stream op; capture-safe.
    (void)hipFuncSetAttribute((const void*)edge_msg_kernel,
                              hipFuncAttributeMaxDynamicSharedMemorySize, SMEM_BYTES);
    (void)hipFuncSetAttribute((const void*)node_combine_kernel,
                              hipFuncAttributeMaxDynamicSharedMemorySize, SMEM_BYTES);

    // 1) zero the segment-sum accumulator
    {
        const long long n = (long long)NVAR * NDIM;
        const int grid = (int)((n + 255) / 256);
        zero_f32_kernel<<<grid, 256, 0, stream>>>(agg, n);
    }
    // 2) edge MLP + scatter-add
    {
        const int grid = (E + TILES_PER_BLOCK * 16 - 1) / (TILES_PER_BLOCK * 16);
        edge_msg_kernel<<<grid, 256, SMEM_BYTES, stream>>>(
            variables, factors, senders, receivers, Wm, bm, agg, E);
    }
    // 3) combine MLP + residual
    {
        const int grid = (NVAR + TILES_PER_BLOCK * 16 - 1) / (TILES_PER_BLOCK * 16);
        node_combine_kernel<<<grid, 256, SMEM_BYTES, stream>>>(
            variables, agg, Wc, bc, out, NVAR);
    }
}
